// ExtractionModel_49014166782558
// MI455X (gfx1250) — compile-verified, matrix operand order checked by reading, not run on hardware
//
#include <hip/hip_runtime.h>
#include <cstdint>

#define HH   128
#define WWID 128
#define HW   16384
#define CCH  512
#define NTOT 49152      // 3 * 16384 candidates
#define NPAD 65536      // pow2 pad for bitonic sort
#define NMAX 2048       // max_features

typedef __attribute__((ext_vector_type(2)))  float    v2f;
typedef __attribute__((ext_vector_type(8)))  float    v8f;
typedef __attribute__((ext_vector_type(16))) _Float16 v16h;

// ---------------- NMS: dets[g] = (3x3 localmax == s) ? s : 0 ----------------
__global__ void em_nms_kernel(const float* __restrict__ s0,
                              const float* __restrict__ s1,
                              const float* __restrict__ s2,
                              float* __restrict__ dets) {
  int gid = blockIdx.x * blockDim.x + threadIdx.x;
  if (gid >= NTOT) return;
  int sc = gid >> 14, pix = gid & (HW - 1);
  int i = pix >> 7, j = pix & 127;
  const float* s = (sc == 0) ? s0 : (sc == 1) ? s1 : s2;
  float c = s[pix];
  float m = -INFINITY;
  #pragma unroll
  for (int dy = -1; dy <= 1; ++dy)
    #pragma unroll
    for (int dx = -1; dx <= 1; ++dx) {
      int y = i + dy, x = j + dx;
      if (y >= 0 && y < HH && x >= 0 && x < WWID) m = fmaxf(m, s[y * WWID + x]);
    }
  dets[gid] = (m == c) ? c : 0.0f;
}

// ------------- sub-pixel displacement + validity; ki,kj,sc to ws ------------
__device__ __forceinline__ float em_pad(const float* m, int y, int x) {
  return (y >= 0 && y < HH && x >= 0 && x < WWID) ? m[y * WWID + x] : 0.0f;
}

__global__ void em_disp_kernel(const float* __restrict__ dets,
                               float* __restrict__ wsc,
                               float* __restrict__ wki,
                               float* __restrict__ wkj) {
  int gid = blockIdx.x * blockDim.x + threadIdx.x;
  if (gid >= NTOT) return;
  int sc = gid >> 14, pix = gid & (HW - 1);
  int i = pix >> 7, j = pix & 127;
  const float* nm = dets + sc * HW;
  float s     = nm[pix];
  float up    = em_pad(nm, i - 1, j);
  float down  = em_pad(nm, i + 1, j);
  float left  = em_pad(nm, i, j - 1);
  float right = em_pad(nm, i, j + 1);
  float ul    = em_pad(nm, i - 1, j - 1);
  float ur    = em_pad(nm, i - 1, j + 1);
  float dl    = em_pad(nm, i + 1, j - 1);
  float dr    = em_pad(nm, i + 1, j + 1);
  float dii = up + down - 2.0f * s;
  float djj = left + right - 2.0f * s;
  float dij = 0.25f * (ul + dr - ur - dl);
  float di  = 0.5f * (down - up);
  float dj  = 0.5f * (right - left);
  float det = dii * djj - dij * dij;
  float safe = (det == 0.0f) ? 1.0f : det;
  float disp_i = -(djj * di - dij * dj) / safe;
  float disp_j = -(-dij * di + dii * dj) / safe;
  bool ok = (det != 0.0f);
  float dif = ok ? disp_i : 1.0f;
  float djf = ok ? disp_j : 1.0f;
  float ki = (float)i + dif;
  float kj = (float)j + djf;
  float itf = floorf(ki), jlf = floorf(kj);
  bool inb = (itf >= 0.0f) && (itf <= 126.0f) && (jlf >= 0.0f) && (jlf <= 126.0f);
  bool valid = (s != 0.0f) && (fabsf(dif) < 0.5f) && (fabsf(djf) < 0.5f) && inb;
  wsc[gid] = valid ? s : -1.0f;
  wki[gid] = ki;
  wkj[gid] = kj;
}

// ---------------- sortable keys: (mapped_score << 32) | (~idx) --------------
__global__ void em_keygen_kernel(const float* __restrict__ wsc,
                                 unsigned long long* __restrict__ keys) {
  int i = blockIdx.x * blockDim.x + threadIdx.x;
  if (i >= NPAD) return;
  unsigned long long key = 0ull;
  if (i < NTOT) {
    unsigned b = __float_as_uint(wsc[i]);
    unsigned u = (b & 0x80000000u) ? ~b : (b | 0x80000000u); // monotone map
    key = ((unsigned long long)u << 32) | (unsigned long long)(0xFFFFFFFFu - (unsigned)i);
  }
  keys[i] = key;
}

// ---------------- global bitonic pass (descending overall) ------------------
__global__ void em_bitonic_pass(unsigned long long* __restrict__ keys, int j, int k) {
  int i = blockIdx.x * blockDim.x + threadIdx.x;
  if (i >= NPAD) return;
  int l = i ^ j;
  if (l > i) {
    unsigned long long a = keys[i], b = keys[l];
    bool descBlock = ((i & k) == 0);
    bool doSwap = descBlock ? (a < b) : (a > b);
    if (doSwap) { keys[i] = b; keys[l] = a; }
  }
}

// ---------------- emit kps (x=ku_j, y=ku_i) and top scores ------------------
__global__ void em_topout_kernel(const unsigned long long* __restrict__ keys,
                                 const float* __restrict__ wki,
                                 const float* __restrict__ wkj,
                                 float* __restrict__ kps,
                                 float* __restrict__ tops) {
  int r = blockIdx.x * blockDim.x + threadIdx.x;
  if (r >= NMAX) return;
  unsigned long long key = keys[r];
  unsigned lo = (unsigned)(key & 0xFFFFFFFFull);
  unsigned idx = 0xFFFFFFFFu - lo;
  if (idx >= NTOT) idx = NTOT - 1;  // safety (padding keys never win in practice)
  unsigned uhi = (unsigned)(key >> 32);
  float scv = (uhi & 0x80000000u) ? __uint_as_float(uhi & 0x7FFFFFFFu)
                                  : __uint_as_float(~uhi);
  float ku_i = wki[idx];
  float ku_j = wkj[idx];
  #pragma unroll
  for (int t = 0; t < 4; ++t) { ku_i = ku_i * 2.0f + 0.5f; ku_j = ku_j * 2.0f + 0.5f; }
  kps[2 * r + 0] = ku_j;
  kps[2 * r + 1] = ku_i;
  tops[r] = scv;
}

// -------- per-keypoint bilinear descriptor via WMMA + L2 normalize ----------
// One wave per selected keypoint. For each 16-channel chunk:
//   D = A(16x4) * B(4x16), A rows all = (w00,w01,w10,w11), B rows = 4 corners.
// f32 WMMA layout (wave32): A: lanes0-15 hold K=0,1; lanes16-31 hold K=2,3.
//                           B: VGPR0/1 = K0/K1 (lanes0-15), K2/K3 (lanes16-31).
__global__ void __launch_bounds__(256)
em_desc_kernel(const float* __restrict__ early,
               const float* __restrict__ middle,
               const float* __restrict__ deep,
               const unsigned long long* __restrict__ keys,
               const float* __restrict__ wki,
               const float* __restrict__ wkj,
               float* __restrict__ descs) {
  int tid  = blockIdx.x * blockDim.x + threadIdx.x;
  int rank = tid >> 5;          // wave id == keypoint rank (grid = 2048 waves)
  int lane = tid & 31;
  if (rank >= NMAX) return;     // uniform per wave; grid sized exactly anyway

  unsigned long long key = keys[rank];
  unsigned idx = 0xFFFFFFFFu - (unsigned)(key & 0xFFFFFFFFull);
  if (idx >= NTOT) idx = NTOT - 1;
  int scl = idx >> 14;
  const float* F = (scl == 0) ? early : (scl == 1) ? middle : deep;

  float ki = wki[idx], kj = wkj[idx];
  float itf = floorf(ki), jlf = floorf(kj);
  float wi = ki - itf, wj = kj - jlf;               // weights from UNclipped floor
  int it = (int)fminf(fmaxf(itf, 0.0f), 126.0f);
  int jl = (int)fminf(fmaxf(jlf, 0.0f), 126.0f);
  float w00 = (1.0f - wi) * (1.0f - wj);
  float w01 = (1.0f - wi) * wj;
  float w10 = wi * (1.0f - wj);
  float w11 = wi * wj;

  int o00 = it * WWID + jl;
  int o01 = o00 + 1;
  int o10 = o00 + WWID;
  int o11 = o10 + 1;

  bool loHalf = (lane < 16);
  int off0 = loHalf ? o00 : o10;   // K=0 / K=2 corner for this lane half
  int off1 = loHalf ? o01 : o11;   // K=1 / K=3 corner for this lane half

  float acc[32];
  float sumsq = 0.0f;

  #pragma unroll
  for (int chunk = 0; chunk < 32; ++chunk) {
    int ch = chunk * 16 + (lane & 15);
    const float* col = F + ch * HW;
    // prefetch next chunk's strided gather lines (emits global_prefetch_b8)
    int chn = ((chunk + 1) & 31) * 16 + (lane & 15);
    __builtin_prefetch(F + chn * HW + off0, 0, 0);

    v2f b; b.x = col[off0]; b.y = col[off1];
    v2f a; a.x = loHalf ? w00 : w10; a.y = loHalf ? w01 : w11;
    v8f c = {};
#if defined(__has_builtin) && __has_builtin(__builtin_amdgcn_wmma_f32_16x16x4_f32)
    v8f d = __builtin_amdgcn_wmma_f32_16x16x4_f32(
        /*neg_a=*/false, a, /*neg_b=*/false, b,
        /*c_mod=*/(short)0, c, /*reuse_a=*/false, /*reuse_b=*/false);
#else
    // fallback: confirmed f16 WMMA (K=32); weights/corners in K=0..3, rest 0
    v16h a16 = {}; v16h b16 = {};
    if (loHalf) {
      a16[0] = (_Float16)w00; a16[1] = (_Float16)w01;
      a16[2] = (_Float16)w10; a16[3] = (_Float16)w11;
      b16[0] = (_Float16)col[o00]; b16[1] = (_Float16)col[o01];
      b16[2] = (_Float16)col[o10]; b16[3] = (_Float16)col[o11];
    }
    v8f d = __builtin_amdgcn_wmma_f32_16x16x32_f16(
        false, a16, false, b16, (short)0, c, false, false);
#endif
    float val = d[0];                         // row0 (rows identical): desc[ch]
    acc[chunk] = val;
    sumsq += loHalf ? val * val : 0.0f;       // lanes 16-31 are duplicates
  }

  // wave reduction of sum of squares (wave32)
  #pragma unroll
  for (int m = 16; m >= 1; m >>= 1) sumsq += __shfl_xor(sumsq, m, 32);
  float inv = 1.0f / fmaxf(sqrtf(sumsq), 1e-12f);

  float* drow = descs + (size_t)rank * CCH;
  #pragma unroll
  for (int chunk = 0; chunk < 32; ++chunk) {
    if (loHalf) drow[chunk * 16 + (lane & 15)] = acc[chunk] * inv;
  }
}

// --------------------------------- host -------------------------------------
extern "C" void kernel_launch(void* const* d_in, const int* in_sizes, int n_in,
                              void* d_out, int out_size, void* d_ws, size_t ws_size,
                              hipStream_t stream) {
  const float* early  = (const float*)d_in[0];
  const float* middle = (const float*)d_in[1];
  const float* deep   = (const float*)d_in[2];
  const float* se     = (const float*)d_in[3];
  const float* sm     = (const float*)d_in[4];
  const float* sd     = (const float*)d_in[5];

  float* out   = (float*)d_out;
  float* kps   = out;                       // 2048*2
  float* descs = out + 2 * NMAX;            // 2048*512
  float* tops  = descs + (size_t)NMAX * CCH;// 2048
  float* dets  = tops + NMAX;               // 3*16384

  unsigned long long* keys = (unsigned long long*)d_ws;       // NPAD * 8B
  float* wsc = (float*)((char*)d_ws + (size_t)NPAD * 8);      // NTOT
  float* wki = wsc + NTOT;                                    // NTOT
  float* wkj = wki + NTOT;                                    // NTOT

  em_nms_kernel<<<(NTOT + 255) / 256, 256, 0, stream>>>(se, sm, sd, dets);
  em_disp_kernel<<<(NTOT + 255) / 256, 256, 0, stream>>>(dets, wsc, wki, wkj);
  em_keygen_kernel<<<NPAD / 256, 256, 0, stream>>>(wsc, keys);

  for (int k = 2; k <= NPAD; k <<= 1)
    for (int j = k >> 1; j >= 1; j >>= 1)
      em_bitonic_pass<<<NPAD / 256, 256, 0, stream>>>(keys, j, k);

  em_topout_kernel<<<NMAX / 256, 256, 0, stream>>>(keys, wki, wkj, kps, tops);
  // one wave (32 threads) per keypoint: 2048 waves = 65536 threads
  em_desc_kernel<<<(NMAX * 32) / 256, 256, 0, stream>>>(
      early, middle, deep, keys, wki, wkj, descs);
}
